// TGCNGraphConvolution_2_52604759441743
// MI455X (gfx1250) — matrix operand order repaired and checked by reading, not compiled, not run on hardware
//
#include <hip/hip_runtime.h>
#include <hip/hip_bf16.h>

// Problem constants (match reference)
#define Bc   8
#define Nc   2048
#define Gc   64
#define CC   65      // G+1 feature columns
#define CP   80      // padded to 5 WMMA N-tiles of 16
#define OUTC 64

typedef __attribute__((ext_vector_type(16))) __bf16 v16bf;
typedef __attribute__((ext_vector_type(8)))  float  v8f;

// ---------------------------------------------------------------------------
// Kernel 1: inv[b,i] = dinv_sqrt^2 with reference's isinf->0 guard.
// d = 1 + sum_j adj[b,i,j]   (the +1 is the identity self-loop)
// One 256-thread block per row; warms adj into L2 for the GEMM pass.
// ---------------------------------------------------------------------------
__global__ __launch_bounds__(256) void k_rowsum(const float* __restrict__ adj,
                                                float* __restrict__ inv) {
    __shared__ float red[256];
    const int row = blockIdx.x;                       // 0 .. B*N-1
    const float4* rp = (const float4*)(adj + (size_t)row * Nc);
    const int t = threadIdx.x;
    float4 x = rp[t];
    float4 y = rp[t + 256];
    float s = x.x + x.y + x.z + x.w + y.x + y.y + y.z + y.w;
    red[t] = s;
    __syncthreads();
    for (int off = 128; off > 0; off >>= 1) {
        if (t < off) red[t] += red[t + off];
        __syncthreads();
    }
    if (t == 0) {
        float d = 1.0f + red[0];
        float r = rsqrtf(d);
        if (isinf(r)) r = 0.0f;                       // power(0,-0.5)=inf -> 0
        inv[row] = r * r;
    }
}

// ---------------------------------------------------------------------------
// Kernel 2: build scaledConcat, TRANSPOSED + bf16 + zero-padded:
//   scBT[b, c, j] = concat[b, j, c] * inv[b, j]   (c in [0,65), zeros to 80)
// Column-major storage means a WMMA B-fragment lane reads 16 contiguous bf16.
// ---------------------------------------------------------------------------
__global__ __launch_bounds__(256) void k_build(const float* __restrict__ inputs,
                                               const float* __restrict__ hidden,
                                               const float* __restrict__ inv,
                                               unsigned short* __restrict__ scBT_u16) {
    __bf16* scBT = (__bf16*)scBT_u16;
    const int idx = blockIdx.x * 256 + threadIdx.x;   // 0 .. B*N-1
    const int b = idx >> 11;
    const int j = idx & (Nc - 1);
    const float s = inv[idx];
    __bf16* base = scBT + (size_t)b * CP * Nc + j;
    base[0] = (__bf16)(inputs[idx] * s);
    const float* h = hidden + (size_t)idx * Gc;
#pragma unroll
    for (int g = 0; g < Gc; ++g)
        base[(size_t)(g + 1) * Nc] = (__bf16)(h[g] * s);
#pragma unroll
    for (int c = CC; c < CP; ++c)
        base[(size_t)c * Nc] = (__bf16)0.0f;
}

// ---------------------------------------------------------------------------
// Kernel 3: tmp[b,i,c] = sum_j adj[b,i,j]*scBT[b,c,j] + concat[b,i,c]*inv[b,i]
// bf16 WMMA 16x16x32, fp32 accumulate. Per-wave K-loop:
//  A fragment: lane group lo/hi holds 16 contiguous floats of its row
//  (K = h*16..h*16+15), matching B, converted
//  fp32->bf16 in-register.
//  B fragment: lanes 0-15 hold K=0..15, lanes 16-31 K=16..31 for column
//  N=lane&15 -> one contiguous 32-byte read from the transposed scBT.
// ---------------------------------------------------------------------------
template <int NT>
__device__ inline void gemm_tiles(const float* __restrict__ A,
                                  const __bf16* __restrict__ Bb,
                                  int nt0, int laneLo, int laneHi,
                                  v8f acc[NT]) {
    for (int k0 = 0; k0 < Nc; k0 += 32) {
        const float* ap = A + k0 + laneHi * 16;
        float4 f0 = *(const float4*)(ap);
        float4 f1 = *(const float4*)(ap + 4);
        float4 f2 = *(const float4*)(ap + 8);
        float4 f3 = *(const float4*)(ap + 12);
        v16bf a;
        a[0]  = (__bf16)f0.x; a[1]  = (__bf16)f0.y; a[2]  = (__bf16)f0.z; a[3]  = (__bf16)f0.w;
        a[4]  = (__bf16)f1.x; a[5]  = (__bf16)f1.y; a[6]  = (__bf16)f1.z; a[7]  = (__bf16)f1.w;
        a[8]  = (__bf16)f2.x; a[9]  = (__bf16)f2.y; a[10] = (__bf16)f2.z; a[11] = (__bf16)f2.w;
        a[12] = (__bf16)f3.x; a[13] = (__bf16)f3.y; a[14] = (__bf16)f3.z; a[15] = (__bf16)f3.w;
#pragma unroll
        for (int t = 0; t < NT; ++t) {
            const int col = (nt0 + t) * 16 + laneLo;
            v16bf bv = *(const v16bf*)(Bb + (size_t)col * Nc + k0 + laneHi * 16);
            acc[t] = __builtin_amdgcn_wmma_f32_16x16x32_bf16(
                false, a, false, bv, (short)0, acc[t], false, false);
        }
    }
}

template <int NT>
__device__ inline void epilogue_tiles(const v8f acc[NT], int nt0,
                                      int b, int rowBase, int laneLo, int laneHi,
                                      const float* __restrict__ inputs,
                                      const float* __restrict__ hidden,
                                      const float* __restrict__ inv,
                                      float* __restrict__ tmp) {
#pragma unroll
    for (int t = 0; t < NT; ++t) {
        const int col = (nt0 + t) * 16 + laneLo;
        if (col < CC) {
#pragma unroll
            for (int r = 0; r < 8; ++r) {
                const int row = rowBase + laneHi * 8 + r;      // C/D layout
                const size_t bi = (size_t)b * Nc + row;
                float cv = (col == 0) ? inputs[bi]
                                      : hidden[bi * Gc + (col - 1)];
                tmp[bi * CP + col] = acc[t][r] + cv * inv[bi];
            }
        }
    }
}

__global__ __launch_bounds__(256) void k_gemm(const float* __restrict__ adj,
                                              const float* __restrict__ inputs,
                                              const float* __restrict__ hidden,
                                              const float* __restrict__ inv,
                                              const unsigned short* __restrict__ scBT_u16,
                                              float* __restrict__ tmp) {
    const __bf16* scBT = (const __bf16*)scBT_u16;
    const int b      = blockIdx.y;
    const int lane   = threadIdx.x & 31;
    const int wave   = threadIdx.x >> 5;     // 0..7
    const int rowGrp = wave & 3;             // 4 row groups of 16 rows
    const int colHalf = wave >> 2;           // 0 -> tiles 0..2, 1 -> tiles 3..4
    const int laneLo = lane & 15;
    const int laneHi = lane >> 4;
    const int rowBase = blockIdx.x * 64 + rowGrp * 16;
    const int aRow = rowBase + laneLo;

    const float*  A  = adj  + ((size_t)b * Nc + aRow) * Nc;
    const __bf16* Bb = scBT + (size_t)b * CP * Nc;

    if (colHalf == 0) {
        v8f acc[3] = {};
        gemm_tiles<3>(A, Bb, 0, laneLo, laneHi, acc);
        epilogue_tiles<3>(acc, 0, b, rowBase, laneLo, laneHi, inputs, hidden, inv, tmp);
    } else {
        v8f acc[2] = {};
        gemm_tiles<2>(A, Bb, 3, laneLo, laneHi, acc);
        epilogue_tiles<2>(acc, 3, b, rowBase, laneLo, laneHi, inputs, hidden, inv, tmp);
    }
}

// ---------------------------------------------------------------------------
// Kernel 4: faithful transpose(2,0,1).reshape scramble + [.,65]@[65,64]+bias.
// scram[r,c] = tmp[b',n',g'] with flat = r*65+c, g'=flat>>14, b'=(flat>>11)&7,
// n'=flat&2047. 4 output rows per 256-thread block; W/bias L2-resident.
// ---------------------------------------------------------------------------
__global__ __launch_bounds__(256) void k_out(const float* __restrict__ tmp,
                                             const float* __restrict__ W,
                                             const float* __restrict__ bias,
                                             float* __restrict__ out) {
    __shared__ float scram[4][CC];
    const int grp = threadIdx.x >> 6;          // row within block: 0..3
    const int o   = threadIdx.x & 63;          // output feature
    const int r   = blockIdx.x * 4 + grp;      // 0 .. B*N-1
    for (int c = o; c < CC; c += 64) {
        const unsigned int flat = (unsigned int)r * CC + c;
        const int g  = flat >> 14;             // / (B*N)
        const int bb = (flat >> 11) & 7;       // / N  % B
        const int nn = flat & (Nc - 1);        // % N
        scram[grp][c] = tmp[((size_t)bb * Nc + nn) * CP + g];
    }
    __syncthreads();
    float s = bias[o];
#pragma unroll
    for (int c = 0; c < CC; ++c)
        s += scram[grp][c] * W[c * OUTC + o];
    out[(size_t)r * OUTC + o] = s;
}

// ---------------------------------------------------------------------------
// Launch. Workspace layout (bytes):
//   [0,                65536)            inv   : float[B*N]
//   [65536,            65536+5242880)    tmp   : float[B*N*80]
//   [5308416,          5308416+2621440)  scBT  : bf16 [B*80*N]
// Total ~7.93 MB.
// ---------------------------------------------------------------------------
extern "C" void kernel_launch(void* const* d_in, const int* in_sizes, int n_in,
                              void* d_out, int out_size, void* d_ws, size_t ws_size,
                              hipStream_t stream) {
    const float* inputs = (const float*)d_in[0];   // [B, N]
    const float* hidden = (const float*)d_in[1];   // [B, N*GRU]
    const float* adj    = (const float*)d_in[2];   // [B, N, N]
    const float* W      = (const float*)d_in[3];   // [65, 64]
    const float* bias   = (const float*)d_in[4];   // [64]
    float* out = (float*)d_out;                    // [B, N*OUT]

    char* ws = (char*)d_ws;
    float*          inv  = (float*)ws;
    float*          tmp  = (float*)(ws + 65536);
    unsigned short* scBT = (unsigned short*)(ws + 65536 + (size_t)Bc * Nc * CP * 4);

    // 1) row sums -> inverse degree (also warms adj into the 192MB L2)
    k_rowsum<<<Bc * Nc, 256, 0, stream>>>(adj, inv);
    // 2) scaled concat, transposed + bf16 + padded
    k_build<<<(Bc * Nc) / 256, 256, 0, stream>>>(inputs, hidden, inv, scBT);
    // 3) batched WMMA GEMM + fused identity term
    dim3 g3(Nc / 64, Bc);
    k_gemm<<<g3, 256, 0, stream>>>(adj, inputs, hidden, inv, scBT, tmp);
    // 4) scramble + small dense layer
    k_out<<<(Bc * Nc) / 4, 256, 0, stream>>>(tmp, W, bias, out);
}
